// SlidingWindowAttention_77189152244391
// MI455X (gfx1250) — compile-verified, matrix operand order checked by reading, not run on hardware
//
#include <hip/hip_runtime.h>
#include <hip/hip_bf16.h>

// ---------------------------------------------------------------------------
// Problem constants (match reference setup_inputs)
// ---------------------------------------------------------------------------
#define BB    4
#define NN    2048
#define DIM   1024
#define HEADS 16
#define DH    64
#define NPM   4
#define DI    1024          // heads*dim_head
#define KVLEN 2052          // NPM + NN
#define KVP   2112          // padded to 33 tiles of 64
#define ROWS  (BB*NN)       // 8192
#define EPSF  1.1920929e-07f

typedef __attribute__((ext_vector_type(16))) __bf16 v16bf;
typedef __attribute__((ext_vector_type(8)))  float  v8f;

union FragU { v16bf v; uint4 q[2]; };

// Load a 16-element bf16 fragment as two 16B chunks (ISA 16-bit A/B layout:
// per lane, VGPR0-3 hold one 8-elem K chunk, VGPR4-7 the other).
__device__ __forceinline__ v16bf ldfrag(const __bf16* p0, const __bf16* p1) {
    FragU f;
    f.q[0] = *reinterpret_cast<const uint4*>(p0);
    f.q[1] = *reinterpret_cast<const uint4*>(p1);
    return f.v;
}

// DPP lane permute (VALU, stays within a 16-lane DPP row == C-layout half).
#define DPPF(x, ctrl) \
    __int_as_float(__builtin_amdgcn_update_dpp(0, __float_as_int(x), (ctrl), 0xf, 0xf, true))

// 16-lane max reduction, all-VALU (no DS traffic):
//  xor1, xor2 -> quad reduced; half-mirror pairs quads; mirror pairs 8-groups.
__device__ __forceinline__ float redmax16(float t) {
    t = fmaxf(t, DPPF(t, 0xB1));   // quad_perm(1,0,3,2)
    t = fmaxf(t, DPPF(t, 0x4E));   // quad_perm(2,3,0,1)
    t = fmaxf(t, DPPF(t, 0x141));  // row_half_mirror
    t = fmaxf(t, DPPF(t, 0x140));  // row_mirror
    return t;
}

// ---------------------------------------------------------------------------
// 1) RMSNorm + cast to bf16.  One block per row of x.
// ---------------------------------------------------------------------------
__global__ void rmsnorm_cast(const float* __restrict__ x,
                             const float* __restrict__ w,
                             __bf16* __restrict__ out) {
    const int row = blockIdx.x;
    const float* xr = x + (size_t)row * DIM;
    float ss = 0.f;
    for (int i = threadIdx.x; i < DIM; i += blockDim.x) {
        float v = xr[i];
        ss += v * v;
    }
    __shared__ float red[256];
    red[threadIdx.x] = ss;
    __syncthreads();
    for (int s = 128; s > 0; s >>= 1) {
        if (threadIdx.x < s) red[threadIdx.x] += red[threadIdx.x + s];
        __syncthreads();
    }
    const float inv = rsqrtf(red[0] * (1.0f / DIM) + EPSF);
    for (int i = threadIdx.x; i < DIM; i += blockDim.x)
        out[(size_t)row * DIM + i] = (__bf16)(xr[i] * inv * w[i]);
}

// ---------------------------------------------------------------------------
// 2) f32 -> bf16 weight conversion
// ---------------------------------------------------------------------------
__global__ void cast_f32_bf16(const float* __restrict__ in,
                              __bf16* __restrict__ out, int n) {
    int i = blockIdx.x * blockDim.x + threadIdx.x;
    if (i < n) out[i] = (__bf16)in[i];
}

// ---------------------------------------------------------------------------
// 3/7) WMMA bf16 GEMM:  C[M,N] = A[M,K] * B[N,K]^T
//      Wave tile 32x64 (2 A-frags reuse 4 B-frags -> 8 WMMA per K-step),
//      block tile 64x256, 8 waves.
// ---------------------------------------------------------------------------
template <bool OUT_BF16>
__global__ void __launch_bounds__(256)
gemm_bf16_wmma(const __bf16* __restrict__ A,
               const __bf16* __restrict__ B,
               void* __restrict__ C,
               int M, int N, int K) {
    const int lane = threadIdx.x & 31;
    const int wave = threadIdx.x >> 5;       // 0..7
    const int l15  = lane & 15;
    const int hh   = lane >> 4;              // K-half select per ISA layout

    const int m0 = blockIdx.x * 64  + (wave & 1) * 32;
    const int n0 = blockIdx.y * 256 + (wave >> 1) * 64;

    v8f acc[2][4] = {};
    const __bf16* arow0 = A + (size_t)(m0 + l15) * K;
    const __bf16* arow1 = A + (size_t)(m0 + 16 + l15) * K;
    const __bf16* brow0 = B + (size_t)(n0 + l15) * K;

    for (int k0 = 0; k0 < K; k0 += 32) {
        __builtin_prefetch(arow0 + k0 + 256, 0, 0);    // global_prefetch_b8
        v16bf af0 = ldfrag(arow0 + k0 + hh * 8, arow0 + k0 + 16 + hh * 8);
        v16bf af1 = ldfrag(arow1 + k0 + hh * 8, arow1 + k0 + 16 + hh * 8);
#pragma unroll
        for (int nt = 0; nt < 4; ++nt) {
            const __bf16* br = brow0 + (size_t)(nt * 16) * K + k0;
            v16bf bf = ldfrag(br + hh * 8, br + 16 + hh * 8);
            acc[0][nt] = __builtin_amdgcn_wmma_f32_16x16x32_bf16(
                false, af0, false, bf, (short)0, acc[0][nt], false, false);
            acc[1][nt] = __builtin_amdgcn_wmma_f32_16x16x32_bf16(
                false, af1, false, bf, (short)0, acc[1][nt], false, false);
        }
    }
    // C layout: VGPR r -> row r (lanes 0-15) / row 8+r (lanes 16-31), col = lane&15
#pragma unroll
    for (int mt = 0; mt < 2; ++mt) {
#pragma unroll
        for (int nt = 0; nt < 4; ++nt) {
            const int col = n0 + nt * 16 + l15;
#pragma unroll
            for (int r = 0; r < 8; ++r) {
                const int row = m0 + mt * 16 + r + (hh ? 8 : 0);
                const float val = acc[mt][nt][r];
                if (OUT_BF16)
                    ((__bf16*)C)[(size_t)row * N + col] = (__bf16)val;
                else
                    ((float*)C)[(size_t)row * N + col] = val;
            }
        }
    }
}

// ---------------------------------------------------------------------------
// 4) RoPE + scatter: qkv[row=b*NN+n, 3*DI] (bf16) ->
//      q  [b,h,n,64]   * 0.125 (softmax scale folded in; exact in bf16)
//      k  [b,h,kv,64]
//      vT [b,h,64,KVP]
//    one thread per (b,h,n,pair)
// ---------------------------------------------------------------------------
__global__ void rope_scatter(const __bf16* __restrict__ qkv,
                             __bf16* __restrict__ qb,
                             __bf16* __restrict__ kb,
                             __bf16* __restrict__ vT) {
    const int idx = blockIdx.x * blockDim.x + threadIdx.x;   // 22 bits
    const int p  = idx & 31;
    const int n  = (idx >> 5) & 2047;
    const int h  = (idx >> 16) & 15;
    const int bi = idx >> 20;

    const size_t row = (size_t)(bi * NN + n) * (3 * DI);
    const int cbase = h * DH + 2 * p;

    const float q0 = (float)qkv[row + cbase];
    const float q1 = (float)qkv[row + cbase + 1];
    const float k0 = (float)qkv[row + DI + cbase];
    const float k1 = (float)qkv[row + DI + cbase + 1];
    const float v0 = (float)qkv[row + 2 * DI + cbase];
    const float v1 = (float)qkv[row + 2 * DI + cbase + 1];

    // inv_freq = theta^(-2p/64); angle = n * inv_freq
    const float ang = (float)n * __powf(10000.0f, -(2.0f * (float)p) / 64.0f);
    float s, c;
    __sincosf(ang, &s, &c);

    const float qscale = 0.125f;          // dim_head^-0.5, folded into Q
    const size_t bh = (size_t)(bi * HEADS + h);
    __bf16* qd = qb + (bh * NN + n) * DH + 2 * p;
    qd[0] = (__bf16)((q0 * c - q1 * s) * qscale);
    qd[1] = (__bf16)((q1 * c + q0 * s) * qscale);
    __bf16* kd = kb + (bh * KVP + (n + NPM)) * DH + 2 * p;
    kd[0] = (__bf16)(k0 * c - k1 * s);
    kd[1] = (__bf16)(k1 * c + k0 * s);
    vT[bh * (size_t)DH * KVP + (size_t)(2 * p)     * KVP + (n + NPM)] = (__bf16)v0;
    vT[bh * (size_t)DH * KVP + (size_t)(2 * p + 1) * KVP + (n + NPM)] = (__bf16)v1;
}

// ---------------------------------------------------------------------------
// 5) persistent-memory rows (kv 0..3) + zero padding (kv 2052..2111)
// ---------------------------------------------------------------------------
__global__ void fill_kv(const float* __restrict__ pm,
                        __bf16* __restrict__ kb,
                        __bf16* __restrict__ vT) {
    const int idx  = blockIdx.x * blockDim.x + threadIdx.x;  // 18 bits
    const int d    = idx & 63;
    const int slot = (idx >> 6) & 63;
    const int h    = (idx >> 12) & 15;
    const int bi   = idx >> 16;
    const int kvr  = (slot < NPM) ? slot : (2048 + slot);    // 0..3 or 2052..2111
    const float val = (slot < NPM) ? pm[(size_t)(h * NPM + slot) * DH + d] : 0.0f;
    const size_t bh = (size_t)(bi * HEADS + h);
    kb[(bh * KVP + kvr) * DH + d]                     = (__bf16)val;
    vT[bh * (size_t)DH * KVP + (size_t)d * KVP + kvr] = (__bf16)val;
}

// ---------------------------------------------------------------------------
// 6) Flash attention.  grid=(NN/64, BB*HEADS), block=128 (4 waves).
//    Each wave: 16 q rows.  Row-max via DPP (VALU), row-sum via WMMA with an
//    all-ones B fragment (result lands directly in C-layout), P transposed
//    through a per-wave LDS tile for the PV matmul.
// ---------------------------------------------------------------------------
__global__ void __launch_bounds__(128)
flash_attn(const __bf16* __restrict__ qb,
           const __bf16* __restrict__ kb,
           const __bf16* __restrict__ vT,
           __bf16* __restrict__ attn_out) {
    const int lane = threadIdx.x & 31;
    const int wave = threadIdx.x >> 5;   // 0..3
    const int l15  = lane & 15;
    const int hh   = lane >> 4;
    const int qblk = blockIdx.x;         // 0..31
    const int bh   = blockIdx.y;         // 0..63
    const int h    = bh & 15;
    const int bi   = bh >> 4;
    const int q0   = qblk * 64 + wave * 16;       // this wave's first q row
    const int roff = hh * 8;                      // C-layout row offset per half

    __shared__ __align__(16) __bf16 plds[4][16][64];

    // all-ones B fragment for row-sum WMMA
    v16bf ones;
#pragma unroll
    for (int i = 0; i < 16; ++i) ones[i] = (__bf16)1.0f;

    // Q fragments: A-layout, lane = row, two K-halves per fragment.
    const __bf16* qrow = qb + ((size_t)bh * NN + q0 + l15) * DH;
    const v16bf a0 = ldfrag(qrow + hh * 8,      qrow + 16 + hh * 8);
    const v16bf a1 = ldfrag(qrow + 32 + hh * 8, qrow + 48 + hh * 8);

    v8f o[4] = {};
    float mrow[8], lrow[8];
#pragma unroll
    for (int r = 0; r < 8; ++r) { mrow[r] = -1e30f; lrow[r] = 0.f; }

    // per-wave causal upper tile: largest j needed is (q0+15)+NPM
    const int jt_max = (q0 + 15 + NPM) / 64;          // inclusive
    const __bf16* kbh = kb + (size_t)bh * KVP * DH;
    const __bf16* vbh = vT + (size_t)bh * DH * KVP;

    for (int jt = 0; jt <= jt_max; ++jt) {
        const int j0 = jt * 64;

        // ---- scores: S = Q(16x64) * K^T (4 tiles of 16 kv cols) ----
        v8f s[4];
#pragma unroll
        for (int nt = 0; nt < 4; ++nt) {
            const __bf16* krow = kbh + (size_t)(j0 + nt * 16 + l15) * DH;
            v16bf b0 = ldfrag(krow + hh * 8,      krow + 16 + hh * 8);
            v16bf b1 = ldfrag(krow + 32 + hh * 8, krow + 48 + hh * 8);
            v8f c = {};
            c = __builtin_amdgcn_wmma_f32_16x16x32_bf16(false, a0, false, b0,
                                                        (short)0, c, false, false);
            c = __builtin_amdgcn_wmma_f32_16x16x32_bf16(false, a1, false, b1,
                                                        (short)0, c, false, false);
            s[nt] = c;
        }

        // ---- causal/pad mask: only boundary tiles pay the VALU cost ----
        // (scale already folded into Q at rope time)
        if (j0 + 63 > q0 + NPM || j0 + 63 >= KVLEN) {   // wave-uniform branch
#pragma unroll
            for (int nt = 0; nt < 4; ++nt) {
#pragma unroll
                for (int r = 0; r < 8; ++r) {
                    const int i = q0 + r + roff;
                    const int j = j0 + nt * 16 + l15;
                    if (j > i + NPM || j >= KVLEN) s[nt][r] = -1e30f;
                }
            }
        }

        // ---- row max: all-VALU DPP reduction within 16-lane halves ----
        float corr[8];
#pragma unroll
        for (int r = 0; r < 8; ++r) {
            float t = fmaxf(fmaxf(s[0][r], s[1][r]), fmaxf(s[2][r], s[3][r]));
            t = redmax16(t);
            const float mn = fmaxf(mrow[r], t);
            corr[r] = __expf(mrow[r] - mn);
            mrow[r] = mn;
        }

        // ---- P = exp(S - m), straight into the LDS transpose tile ----
#pragma unroll
        for (int nt = 0; nt < 4; ++nt)
#pragma unroll
            for (int r = 0; r < 8; ++r)
                plds[wave][r + roff][nt * 16 + l15] =
                    (__bf16)__expf(s[nt][r] - mrow[r]);

        const __bf16* pr = &plds[wave][l15][0];
        v16bf pa0 = ldfrag(pr + hh * 8,      pr + 16 + hh * 8);
        v16bf pa1 = ldfrag(pr + 32 + hh * 8, pr + 48 + hh * 8);

        // ---- row sum via WMMA: P(16x64) @ ones -> sums in C layout ----
        v8f rs = {};
        rs = __builtin_amdgcn_wmma_f32_16x16x32_bf16(false, pa0, false, ones,
                                                     (short)0, rs, false, false);
        rs = __builtin_amdgcn_wmma_f32_16x16x32_bf16(false, pa1, false, ones,
                                                     (short)0, rs, false, false);
#pragma unroll
        for (int r = 0; r < 8; ++r) {
            lrow[r] = lrow[r] * corr[r] + rs[r];
#pragma unroll
            for (int nt = 0; nt < 4; ++nt)
                o[nt][r] = o[nt][r] * corr[r];
        }

        // ---- O += P(16x64) * V(64x64); vT gives contiguous B fragments ----
#pragma unroll
        for (int nt = 0; nt < 4; ++nt) {
            const __bf16* vrow = vbh + (size_t)(nt * 16 + l15) * KVP + j0;
            v16bf vb0 = ldfrag(vrow + hh * 8,      vrow + 16 + hh * 8);
            v16bf vb1 = ldfrag(vrow + 32 + hh * 8, vrow + 48 + hh * 8);
            o[nt] = __builtin_amdgcn_wmma_f32_16x16x32_bf16(false, pa0, false, vb0,
                                                            (short)0, o[nt], false, false);
            o[nt] = __builtin_amdgcn_wmma_f32_16x16x32_bf16(false, pa1, false, vb1,
                                                            (short)0, o[nt], false, false);
        }
    }

    // ---- epilogue: multiply by 1/denom, store to [b, n, heads*64] ----
    float inv[8];
#pragma unroll
    for (int r = 0; r < 8; ++r) inv[r] = 1.0f / lrow[r];
#pragma unroll
    for (int nt = 0; nt < 4; ++nt) {
        const int col = h * DH + nt * 16 + l15;
#pragma unroll
        for (int r = 0; r < 8; ++r) {
            const int i = q0 + r + roff;
            attn_out[((size_t)bi * NN + i) * DI + col] = (__bf16)(o[nt][r] * inv[r]);
        }
    }
}

// ---------------------------------------------------------------------------
// launch
// ---------------------------------------------------------------------------
extern "C" void kernel_launch(void* const* d_in, const int* in_sizes, int n_in,
                              void* d_out, int out_size, void* d_ws, size_t ws_size,
                              hipStream_t stream) {
    const float* x    = (const float*)d_in[0];
    const float* nw   = (const float*)d_in[1];
    const float* wqkv = (const float*)d_in[2];
    const float* wout = (const float*)d_in[3];
    const float* pm   = (const float*)d_in[4];
    float* out = (float*)d_out;

    // workspace carve-up (aliased where lifetimes allow)
    char* ws = (char*)d_ws;
    size_t off = 0;
    auto carve = [&](size_t bytes) {
        char* p = ws + off;
        off = (off + bytes + 255) & ~(size_t)255;
        return p;
    };
    const size_t hb_bytes   = (size_t)ROWS * DIM * 2;          // 16 MB
    const size_t qkv_bytes  = (size_t)ROWS * 3 * DI * 2;       // 48 MB
    char* hb_region  = carve(hb_bytes);                        // hb, later qbuf
    __bf16* wqkvb    = (__bf16*)carve((size_t)3 * DI * DIM * 2);
    __bf16* woutb    = (__bf16*)carve((size_t)DIM * DI * 2);
    char* qkv_region = carve(qkv_bytes);                       // qkvb, later attnb
    __bf16* kbuf     = (__bf16*)carve((size_t)BB * HEADS * KVP * DH * 2);
    __bf16* vtbuf    = (__bf16*)carve((size_t)BB * HEADS * DH * KVP * 2);

    __bf16* hb    = (__bf16*)hb_region;    // live: rmsnorm -> gemm1
    __bf16* qbuf  = (__bf16*)hb_region;    // live: rope -> flash (aliases hb)
    __bf16* qkvb  = (__bf16*)qkv_region;   // live: gemm1 -> rope
    __bf16* attnb = (__bf16*)qkv_region;   // live: flash -> gemm2 (aliases qkvb)

    // 1) RMSNorm + cast
    rmsnorm_cast<<<ROWS, 256, 0, stream>>>(x, nw, hb);
    // 2) weight casts
    cast_f32_bf16<<<(3 * DI * DIM + 255) / 256, 256, 0, stream>>>(wqkv, wqkvb, 3 * DI * DIM);
    cast_f32_bf16<<<(DIM * DI + 255) / 256, 256, 0, stream>>>(wout, woutb, DIM * DI);
    // 5) pm + padding (independent of 1-4)
    fill_kv<<<(BB * HEADS * 64 * 64) / 256, 256, 0, stream>>>(pm, kbuf, vtbuf);
    // 3) qkv = h @ w_qkv^T   (8192 x 3072 x 1024)
    gemm_bf16_wmma<true><<<dim3(ROWS / 64, (3 * DI) / 256), 256, 0, stream>>>(
        hb, wqkvb, qkvb, ROWS, 3 * DI, DIM);
    // 4) RoPE + scatter to q/k/vT (scale folded into q)
    rope_scatter<<<(BB * HEADS * NN * 32) / 256, 256, 0, stream>>>(qkvb, qbuf, kbuf, vtbuf);
    // 6) flash attention
    flash_attn<<<dim3(NN / 64, BB * HEADS), 128, 0, stream>>>(qbuf, kbuf, vtbuf, attnb);
    // 7) out = attn @ w_out^T  (8192 x 1024 x 1024), fp32 result
    gemm_bf16_wmma<false><<<dim3(ROWS / 64, DIM / 256), 256, 0, stream>>>(
        attnb, woutb, (void*)out, ROWS, DIM, DI);
}
